// SelfAttention_20512763806242
// MI455X (gfx1250) — matrix operand order repaired and checked by reading, not compiled, not run on hardware
//
#include <hip/hip_runtime.h>

#define BATCH 4
#define SEQ   2048
#define DM    1024
#define MTOT  (BATCH * SEQ)   // 8192 rows of x / q / k / v

typedef __attribute__((ext_vector_type(16))) __bf16 v16bf;
typedef __attribute__((ext_vector_type(8)))  __bf16 v8bf;
typedef __attribute__((ext_vector_type(8)))  float  v8f;

// ---------------------------------------------------------------------------
// Fragment loaders (wave32 WMMA 16x16x32 bf16 layouts, CDNA5 ISA 7.12.2)
// A (16M x 32K), row-major source, leading dim ld:
//   lanes 0-15  : m = lane,    elems 0-7 -> K 0..7,   elems 8-15 -> K 16..23
//   lanes 16-31 : m = lane-16, elems 0-7 -> K 8..15,  elems 8-15 -> K 24..31
// ---------------------------------------------------------------------------
__device__ __forceinline__ v16bf load_a_frag(const __bf16* __restrict__ tile,
                                             int ld, int lane) {
  const int m    = lane & 15;
  const int koff = (lane & 16) ? 8 : 0;
  const __bf16* p = tile + (size_t)m * ld + koff;
  union { v16bf v; v8bf h[2]; } u;
  u.h[0] = *(const v8bf*)(p);        // K = koff .. koff+7
  u.h[1] = *(const v8bf*)(p + 16);   // K = koff+16 .. koff+23
  return u.v;
}

// B (32K x 16N) supplied as Bt[n][k] (K-contiguous), leading dim ld:
//   lanes 0-15  : n = lane,    elems 0-7 -> K 0..7,   elems 8-15 -> K 16..23
//   lanes 16-31 : n = lane-16, elems 0-7 -> K 8..15,  elems 8-15 -> K 24..31
__device__ __forceinline__ v16bf load_b_frag(const __bf16* __restrict__ tile,
                                             int ld, int lane) {
  const int n    = lane & 15;
  const int koff = (lane & 16) ? 8 : 0;
  const __bf16* p = tile + (size_t)n * ld + koff;
  union { v16bf v; v8bf h[2]; } u;
  u.h[0] = *(const v8bf*)(p);
  u.h[1] = *(const v8bf*)(p + 16);
  return u.v;
}

// ---------------------------------------------------------------------------
// Wave-level 64(M) x 32(N) GEMM core: 8 accumulators, 8 WMMAs / K-step of 32.
// Register budget: 8*v8f acc (64 VGPR) + 6 frags (48 VGPR)  -> no spills
// (verified: 16-accumulator variants spill to scratch in the hot loop).
// ---------------------------------------------------------------------------
template <int KDIM>
__device__ __forceinline__ void wave_gemm_64x32(const __bf16* __restrict__ A, int lda,
                                                const __bf16* __restrict__ Bt, int ldb,
                                                int lane, v8f (&acc)[4][2]) {
  #pragma unroll 1
  for (int kb = 0; kb < KDIM; kb += 32) {
    v16bf af[4], bfr[2];
    #pragma unroll
    for (int i = 0; i < 4; ++i)
      af[i] = load_a_frag(A + (size_t)(16 * i) * lda + kb, lda, lane);
    #pragma unroll
    for (int j = 0; j < 2; ++j)
      bfr[j] = load_b_frag(Bt + (size_t)(16 * j) * ldb + kb, ldb, lane);
    #pragma unroll
    for (int i = 0; i < 4; ++i) {
      #pragma unroll
      for (int j = 0; j < 2; ++j) {
        acc[i][j] = __builtin_amdgcn_wmma_f32_16x16x32_bf16(
            false, af[i], false, bfr[j], (short)0, acc[i][j], false, false);
      }
    }
  }
}

// ---------------------------------------------------------------------------
// Prep kernels
// ---------------------------------------------------------------------------
__global__ void k_convert_bf16(const float* __restrict__ in,
                               __bf16* __restrict__ out, int n) {
  int i = (blockIdx.x * blockDim.x + threadIdx.x) * 4;
  if (i + 3 < n) {
    float4 v = *(const float4*)(in + i);
    out[i]     = (__bf16)v.x;
    out[i + 1] = (__bf16)v.y;
    out[i + 2] = (__bf16)v.z;
    out[i + 3] = (__bf16)v.w;
  }
}

// Wt[n][k] = (bf16) W[k][n]   (1024x1024)
__global__ void k_transpose_w(const float* __restrict__ W,
                              __bf16* __restrict__ Wt) {
  int idx = blockIdx.x * blockDim.x + threadIdx.x;  // n*1024 + k
  int nn = idx >> 10;
  int kk = idx & (DM - 1);
  Wt[idx] = (__bf16)W[(size_t)kk * DM + nn];
}

// ---------------------------------------------------------------------------
// QKV projection: out = (Xb @ Wt^T + bias) * alpha  -> bf16
// transposed==0: out[m][n] (ld DM);  transposed==1: out[b][n][s] (V^T, ld SEQ)
// ---------------------------------------------------------------------------
__global__ void k_qkv_gemm(const __bf16* __restrict__ Xb,
                           const __bf16* __restrict__ Wt,
                           const float* __restrict__ bias,
                           __bf16* __restrict__ out,
                           float alpha, int transposed) {
  const int lane  = threadIdx.x;
  const int mbase = blockIdx.x * 64;
  const int nbase = blockIdx.y * 32;

  v8f acc[4][2] = {};
  wave_gemm_64x32<DM>(Xb + (size_t)mbase * DM, DM,
                      Wt + (size_t)nbase * DM, DM, lane, acc);

  const int nl = lane & 15;
  const int mo = (lane & 16) ? 8 : 0;
  #pragma unroll
  for (int i = 0; i < 4; ++i) {
    #pragma unroll
    for (int j = 0; j < 2; ++j) {
      const int c = nbase + 16 * j + nl;
      const float bs = bias[c];
      #pragma unroll
      for (int r = 0; r < 8; ++r) {
        const int m = mbase + 16 * i + mo + r;
        const float v = (acc[i][j][r] + bs) * alpha;
        if (!transposed) {
          out[(size_t)m * DM + c] = (__bf16)v;
        } else {
          const int bat = m >> 11;          // m / SEQ
          const int s   = m & (SEQ - 1);
          out[(size_t)bat * DM * SEQ + (size_t)c * SEQ + s] = (__bf16)v;
        }
      }
    }
  }
}

// ---------------------------------------------------------------------------
// scores: S[b][i][j] = Q[b][i][:] . K[b][j][:]   (scale already folded into Q)
// ---------------------------------------------------------------------------
__global__ void k_scores_gemm(const __bf16* __restrict__ Qb,
                              const __bf16* __restrict__ Kb,
                              float* __restrict__ S) {
  const int lane  = threadIdx.x;
  const int mbase = blockIdx.x * 64;
  const int nbase = blockIdx.y * 32;
  const int bat   = blockIdx.z;

  const __bf16* A  = Qb + (size_t)bat * SEQ * DM + (size_t)mbase * DM;
  const __bf16* Bt = Kb + (size_t)bat * SEQ * DM + (size_t)nbase * DM;

  v8f acc[4][2] = {};
  wave_gemm_64x32<DM>(A, DM, Bt, DM, lane, acc);

  float* Sb = S + (size_t)bat * SEQ * SEQ;
  const int nl = lane & 15;
  const int mo = (lane & 16) ? 8 : 0;
  #pragma unroll
  for (int i = 0; i < 4; ++i)
    #pragma unroll
    for (int j = 0; j < 2; ++j)
      #pragma unroll
      for (int r = 0; r < 8; ++r)
        Sb[(size_t)(mbase + 16 * i + mo + r) * SEQ + (nbase + 16 * j + nl)] =
            acc[i][j][r];
}

// ---------------------------------------------------------------------------
// row softmax: P = softmax(S) -> bf16.  One 256-thread block per row (2048 el).
// ---------------------------------------------------------------------------
__global__ void k_softmax(const float* __restrict__ S, __bf16* __restrict__ P) {
  const int row  = blockIdx.x;                 // 0 .. MTOT-1
  const int tid  = threadIdx.x;                // 0 .. 255
  const int lane = tid & 31;
  const int wid  = tid >> 5;
  const float* sr = S + (size_t)row * SEQ;
  __bf16* pr      = P + (size_t)row * SEQ;

  __shared__ float red[8];

  float v[8];
  float m = -3.0e38f;
  #pragma unroll
  for (int t = 0; t < 8; ++t) {
    v[t] = sr[tid + 256 * t];
    m = fmaxf(m, v[t]);
  }
  #pragma unroll
  for (int o = 16; o > 0; o >>= 1) m = fmaxf(m, __shfl_xor(m, o, 32));
  if (lane == 0) red[wid] = m;
  __syncthreads();
  m = red[lane & 7];
  #pragma unroll
  for (int o = 4; o > 0; o >>= 1) m = fmaxf(m, __shfl_xor(m, o, 32));

  float e[8];
  float s = 0.0f;
  #pragma unroll
  for (int t = 0; t < 8; ++t) {
    e[t] = __expf(v[t] - m);
    s += e[t];
  }
  #pragma unroll
  for (int o = 16; o > 0; o >>= 1) s += __shfl_xor(s, o, 32);
  __syncthreads();                 // protect red[] reuse
  if (lane == 0) red[wid] = s;
  __syncthreads();
  s = red[lane & 7];
  #pragma unroll
  for (int o = 4; o > 0; o >>= 1) s += __shfl_xor(s, o, 32);

  const float inv = 1.0f / s;
  #pragma unroll
  for (int t = 0; t < 8; ++t) pr[tid + 256 * t] = (__bf16)(e[t] * inv);
}

// ---------------------------------------------------------------------------
// out: O[b][i][d] = P[b][i][:] @ V[b][:][d]   (V supplied transposed: Vt[b][d][t])
// ---------------------------------------------------------------------------
__global__ void k_out_gemm(const __bf16* __restrict__ P,
                           const __bf16* __restrict__ Vt,
                           float* __restrict__ O) {
  const int lane  = threadIdx.x;
  const int mbase = blockIdx.x * 64;   // query rows
  const int nbase = blockIdx.y * 32;   // d columns
  const int bat   = blockIdx.z;

  const __bf16* A  = P  + (size_t)bat * SEQ * SEQ + (size_t)mbase * SEQ;
  const __bf16* Bt = Vt + (size_t)bat * DM * SEQ  + (size_t)nbase * SEQ;

  v8f acc[4][2] = {};
  wave_gemm_64x32<SEQ>(A, SEQ, Bt, SEQ, lane, acc);

  float* Ob = O + (size_t)bat * SEQ * DM;
  const int nl = lane & 15;
  const int mo = (lane & 16) ? 8 : 0;
  #pragma unroll
  for (int i = 0; i < 4; ++i)
    #pragma unroll
    for (int j = 0; j < 2; ++j)
      #pragma unroll
      for (int r = 0; r < 8; ++r)
        Ob[(size_t)(mbase + 16 * i + mo + r) * DM + (nbase + 16 * j + nl)] =
            acc[i][j][r];
}

// ---------------------------------------------------------------------------
extern "C" void kernel_launch(void* const* d_in, const int* in_sizes, int n_in,
                              void* d_out, int out_size, void* d_ws, size_t ws_size,
                              hipStream_t stream) {
  (void)in_sizes; (void)n_in; (void)out_size; (void)ws_size;

  const float* x  = (const float*)d_in[0];
  const float* Wq = (const float*)d_in[1];
  const float* bq = (const float*)d_in[2];
  const float* Wk = (const float*)d_in[3];
  const float* bk = (const float*)d_in[4];
  const float* Wv = (const float*)d_in[5];
  const float* bv = (const float*)d_in[6];
  float* out = (float*)d_out;

  char* ws = (char*)d_ws;
  size_t off = 0;
  auto alloc = [&](size_t bytes) -> void* {
    void* p = ws + off;
    off += (bytes + 255) & ~(size_t)255;
    return p;
  };
  __bf16* Xb  = (__bf16*)alloc((size_t)MTOT * DM * 2);        // 16 MB
  __bf16* Wtq = (__bf16*)alloc((size_t)DM * DM * 2);          //  2 MB
  __bf16* Wtk = (__bf16*)alloc((size_t)DM * DM * 2);          //  2 MB
  __bf16* Wtv = (__bf16*)alloc((size_t)DM * DM * 2);          //  2 MB
  __bf16* Qb  = (__bf16*)alloc((size_t)MTOT * DM * 2);        // 16 MB (scale folded)
  __bf16* Kb  = (__bf16*)alloc((size_t)MTOT * DM * 2);        // 16 MB
  __bf16* Vt  = (__bf16*)alloc((size_t)BATCH * DM * SEQ * 2); // 16 MB
  float*  Sc  = (float*) alloc((size_t)BATCH * SEQ * SEQ * 4);// 64 MB
  __bf16* Pp  = (__bf16*)alloc((size_t)BATCH * SEQ * SEQ * 2);// 32 MB

  // 1) precision conversion / weight transpose
  k_convert_bf16<<<(MTOT * DM / 4 + 255) / 256, 256, 0, stream>>>(x, Xb, MTOT * DM);
  k_transpose_w<<<DM * DM / 256, 256, 0, stream>>>(Wq, Wtq);
  k_transpose_w<<<DM * DM / 256, 256, 0, stream>>>(Wk, Wtk);
  k_transpose_w<<<DM * DM / 256, 256, 0, stream>>>(Wv, Wtv);

  // 2) QKV projections (scale 1/sqrt(1024)=1/32 folded into Q; V stored transposed)
  const float scale = 1.0f / 32.0f;
  dim3 gp(MTOT / 64, DM / 32);
  k_qkv_gemm<<<gp, 32, 0, stream>>>(Xb, Wtq, bq, Qb, scale, 0);
  k_qkv_gemm<<<gp, 32, 0, stream>>>(Xb, Wtk, bk, Kb, 1.0f, 0);
  k_qkv_gemm<<<gp, 32, 0, stream>>>(Xb, Wtv, bv, Vt, 1.0f, 1);

  // 3) scores = Q K^T (K row-major already serves as Bt)
  k_scores_gemm<<<dim3(SEQ / 64, SEQ / 32, BATCH), 32, 0, stream>>>(Qb, Kb, Sc);

  // 4) row-wise softmax -> bf16 probabilities
  k_softmax<<<MTOT, 256, 0, stream>>>(Sc, Pp);

  // 5) O = P @ V (via V^T)
  k_out_gemm<<<dim3(SEQ / 64, DM / 32, BATCH), 32, 0, stream>>>(Pp, Vt, out);
}
